// MambaBlock_63118839382421
// MI455X (gfx1250) — compile-verified, matrix operand order checked by reading, not run on hardware
//
#include <hip/hip_runtime.h>
#include <stdint.h>

// ---------------- problem sizes ----------------
#define B_    16
#define CIN   64
#define HIDC  256
#define HH    64
#define WW    64
#define LSEQ  4096            // HH*WW
#define MTOT  65536           // B_*LSEQ tokens

typedef __attribute__((ext_vector_type(16))) __bf16 v16bf;
typedef __attribute__((ext_vector_type(8)))  float  v8f;

union FragU { v16bf v; uint4 q[2]; };

// Load one 16-lane-striped bf16 WMMA fragment (A or B^T), row-major source,
// per-lane: 8 bf16 at [kb] and 8 bf16 at [kb+16].
__device__ inline v16bf ld_frag(const __bf16* p) {
  FragU f;
  f.q[0] = *(const uint4*)(p);
  f.q[1] = *(const uint4*)(p + 16);
  return f.v;
}

// ---------------- async global->LDS staging (ASYNCcnt path) ----------------
#if defined(__AMDGCN__)
#if __has_builtin(__builtin_amdgcn_global_load_async_to_lds_b128)
#define ASYNC_LDS_OK 1
#else
#define ASYNC_LDS_OK 0
#endif
#else
#define ASYNC_LDS_OK 0
#endif

typedef int v4i_t __attribute__((vector_size(16)));
typedef __attribute__((address_space(1))) v4i_t gv4i;   // device/global aperture
typedef __attribute__((address_space(3))) v4i_t lv4i;   // LDS aperture (32-bit)

__device__ inline void cp16_async(void* lds, const void* g) {
#if ASYNC_LDS_OK
  __builtin_amdgcn_global_load_async_to_lds_b128(
      (gv4i*)(uintptr_t)g,
      (lv4i*)(uint32_t)(uintptr_t)lds, 0, 0);
#else
  *(uint4*)lds = *(const uint4*)g;
#endif
}

__device__ inline void async_wait_all() {
#if ASYNC_LDS_OK
#if __has_builtin(__builtin_amdgcn_s_wait_asynccnt)
  __builtin_amdgcn_s_wait_asynccnt(0);
#else
  asm volatile("s_wait_asynccnt 0x0" ::: "memory");
#endif
#endif
}

// ---------------- Tensor Data Mover staging (TENSORcnt path) ----------------
#if defined(__AMDGCN__)
#if __has_builtin(__builtin_amdgcn_tensor_load_to_lds) && __has_builtin(__builtin_amdgcn_s_wait_tensorcnt)
#define TDM_OK 1
#else
#define TDM_OK 0
#endif
#else
#define TDM_OK 0
#endif

#if TDM_OK
typedef unsigned int u32x4 __attribute__((ext_vector_type(4)));
typedef int          i32x4 __attribute__((ext_vector_type(4)));
typedef int          i32x8 __attribute__((ext_vector_type(8)));

// One TDM descriptor load: group0 built from pointers (count=1, type=2 image),
// group1/group2 supplied by caller per ISA section 8.4/8.5 bitfields.
__device__ inline void tdm_load(const void* g, void* lds, i32x8 g1v, i32x4 g2v) {
  uint64_t ga = (uint64_t)(uintptr_t)g;
  u32x4 g0;
  g0[0] = 1u;                                    // count=1 (valid), user mode
  g0[1] = (uint32_t)(uintptr_t)lds;              // lds_addr (LDS byte offset)
  g0[2] = (uint32_t)ga;                          // global_addr[31:0]
  g0[3] = ((uint32_t)(ga >> 32) & 0x01FFFFFFu) | 0x80000000u; // [56:32] | type=2
  i32x4 z4 = {0, 0, 0, 0};
  (void)z4;
#if __clang_major__ >= 23
  i32x8 z8 = {0, 0, 0, 0, 0, 0, 0, 0};
  __builtin_amdgcn_tensor_load_to_lds(g0, g1v, g2v, z4, z8, 0);
#else
  __builtin_amdgcn_tensor_load_to_lds(g0, g1v, g2v, z4, 0);
#endif
}

__device__ inline void tdm_wait_all() {
  __builtin_amdgcn_s_wait_tensorcnt(0);
}
#endif // TDM_OK

// ---------------- prep kernels ----------------
__global__ void cast_f32_bf16_k(const float* __restrict__ s, __bf16* __restrict__ d, int n) {
  int i = blockIdx.x * blockDim.x + threadIdx.x;
  if (i < n) d[i] = (__bf16)s[i];
}

__global__ void nchw2nhwc_k(const float* __restrict__ x, __bf16* __restrict__ xn) {
  int i = blockIdx.x * blockDim.x + threadIdx.x;   // B*L*CIN = 4194304
  if (i >= B_ * LSEQ * CIN) return;
  int c  = i & (CIN - 1);
  int hw = (i >> 6) & (LSEQ - 1);
  int b  = i >> 18;
  xn[i] = (__bf16)x[(((size_t)(b * CIN + c)) << 12) + hw];
}

// Wc1 OIHW (256,64,3,3) f32 -> [tap][O][I] bf16
__global__ void packconv_k(const float* __restrict__ ws, __bf16* __restrict__ wd) {
  int i = blockIdx.x * blockDim.x + threadIdx.x;   // 9*256*64 = 147456
  if (i >= 9 * HIDC * CIN) return;
  int t = i >> 14;
  int r = i & 16383;
  int o = r >> 6;
  int ci = r & 63;
  wd[i] = (__bf16)ws[((size_t)(o * CIN + ci)) * 9 + t];
}

// ---------------- conv 3x3 (implicit GEMM) ----------------
// Halo pixels: async (ASYNCcnt) staging, site stride padded 128B->144B.
// Weights for this block's 64 N-columns: one 3D TDM descriptor
// (dim0=64 k, dim1=64 rows, dim2=9 taps; pad 128B rows by 16B -> [tap][row][72]).
__global__ void __launch_bounds__(256) conv_k(const __bf16* __restrict__ xn,
                                              const __bf16* __restrict__ wc,
                                              const float*  __restrict__ bias,
                                              float* __restrict__ y) {
  __shared__ __bf16 sx[198 * 72];                  // 28512 B halo tile
#if TDM_OK
  __shared__ __bf16 sw[9 * 64 * 72];               // 82944 B weight tile
#endif
  const int tid = threadIdx.x;
  const int bh  = blockIdx.x;                      // b*64 + h
  const int b   = bh >> 6;
  const int h   = bh & 63;
  const int n0  = blockIdx.y * 64;

#if TDM_OK
  if (tid < 32) {                                  // one wave issues the TDM op
    // group1: data_size=2B, pad_enable, pad_interval=32dw(128B), pad_amount=4dw(16B)
    // tensor_dim0=64, tensor_dim1=64, tile_dim0=64, tile_dim1=64, tile_dim2=9,
    // tensor_dim0_stride=64, tensor_dim1_stride=16384 (tap stride)
    i32x8 g1c = {0x07110000, 0x00400000, 0x00400000, 0x00400000,
                 (9 << 16) | 64, 64, 0x40000000, 0};
    i32x4 g2c = {9, 0, 0, 0};                      // tensor_dim2=9
    tdm_load(wc + ((size_t)n0 << 6), sw, g1c, g2c);
  }
#endif

  // interior halo: 3 rows x 512 x 16B chunks (each global row is 8KB contiguous)
  for (int q = tid; q < 1536; q += 256) {
    int dy = q >> 9;
    int e  = q & 511;                              // chunk within row
    int hy = h + dy - 1;
    char* dst = (char*)sx + (dy * 66 + 1 + (e >> 3)) * 144 + (e & 7) * 16;
    if ((unsigned)hy < 64u) {
      const char* src = (const char*)xn + (((size_t)(b * 64 + hy)) << 13) + ((size_t)e << 4);
      cp16_async(dst, src);
    } else {
      *(uint4*)dst = make_uint4(0u, 0u, 0u, 0u);   // zero-pad h edge
    }
  }
  // border columns (wi = 0 and 65) always zero: 3 rows * 2 sites * 8 chunks
  if (tid < 48) {
    int dy = tid / 16;
    int r  = tid & 15;
    int wi = (r >> 3) ? 65 : 0;
    int ci = r & 7;
    *(uint4*)((char*)sx + (dy * 66 + wi) * 144 + ci * 16) = make_uint4(0u, 0u, 0u, 0u);
  }
  async_wait_all();
#if TDM_OK
  if (tid < 32) tdm_wait_all();
#endif
  __syncthreads();

  const int lane = tid & 31;
  const int wv   = tid >> 5;
  const int mw   = (wv & 3) * 16;                  // 4 waves over M=64
  const int nsel = wv >> 2;                        // 2 waves over N=64 (32 each)
  const int lm   = lane & 15;
  const int hi   = lane >> 4;
  const int kb   = hi * 8;

  v8f acc[2];
  #pragma unroll
  for (int nf = 0; nf < 2; ++nf) acc[nf] = v8f{0.f,0.f,0.f,0.f,0.f,0.f,0.f,0.f};

  #pragma unroll
  for (int t = 0; t < 9; ++t) {
    const int dy = t / 3, dx = t % 3;
    #pragma unroll
    for (int ks = 0; ks < 2; ++ks) {
      v16bf a = ld_frag((const __bf16*)((const char*)sx +
                 (dy * 66 + (mw + lm + dx)) * 144 + (ks * 32 + kb) * 2));
      #pragma unroll
      for (int nf = 0; nf < 2; ++nf) {
        const int nl = nsel * 32 + nf * 16 + lm;   // row within weight tile
#if TDM_OK
        v16bf bm = ld_frag((const __bf16*)((const char*)sw +
                    t * 9216 + nl * 144 + (ks * 32 + kb) * 2));
#else
        v16bf bm = ld_frag(wc + (((size_t)t * HIDC + (n0 + nl)) << 6) + ks * 32 + kb);
#endif
        acc[nf] = __builtin_amdgcn_wmma_f32_16x16x32_bf16(false, a, false, bm,
                                                          (short)0, acc[nf], false, false);
      }
    }
  }

  #pragma unroll
  for (int nf = 0; nf < 2; ++nf) {
    const int n = n0 + nsel * 32 + nf * 16 + lm;
    const float bv = bias[n];
    #pragma unroll
    for (int r = 0; r < 8; ++r) {
      int m = mw + r + hi * 8;                     // pixel w within row
      y[(((size_t)bh * 64 + m) << 8) + n] = acc[nf][r] + bv;
    }
  }
}

// ---------------- BatchNorm stats (deterministic 2-stage) ----------------
__global__ void bnstats_k(const float* __restrict__ y, float* __restrict__ part) {
  int blk = blockIdx.x;                            // 256 pixel slabs
  int c = threadIdx.x;
  const float* p = y + ((size_t)blk << 16) + c;
  float s = 0.f, sq = 0.f;
  for (int i = 0; i < 256; ++i) {
    __builtin_prefetch(p + ((size_t)(i + 16) << 8), 0, 0);
    float v = p[(size_t)i << 8];
    s += v; sq += v * v;
  }
  part[(blk << 9) + c] = s;
  part[(blk << 9) + 256 + c] = sq;
}

__global__ void bnfinal_k(const float* __restrict__ part, const float* __restrict__ gamma,
                          const float* __restrict__ beta, float* __restrict__ ss) {
  int c = threadIdx.x;
  float s = 0.f, sq = 0.f;
  for (int i = 0; i < 256; ++i) {
    s  += part[(i << 9) + c];
    sq += part[(i << 9) + 256 + c];
  }
  const float inv_n = 1.f / 65536.f;
  float mean = s * inv_n;
  float var  = sq * inv_n - mean * mean;
  float sc   = gamma[c] * rsqrtf(var + 1e-5f);
  ss[c]       = sc;
  ss[256 + c] = beta[c] - mean * sc;
}

__global__ void bnact_k(const float* __restrict__ y, const float* __restrict__ ss,
                        __bf16* __restrict__ yb) {
  int i = blockIdx.x * blockDim.x + threadIdx.x;   // float4 groups: 4194304
  if (i >= (MTOT * HIDC) / 4) return;
  const float4 v = *(const float4*)(y + ((size_t)i << 2));
  int c = (i << 2) & 255;
  float a0 = v.x * ss[c + 0] + ss[256 + c + 0];
  float a1 = v.y * ss[c + 1] + ss[256 + c + 1];
  float a2 = v.z * ss[c + 2] + ss[256 + c + 2];
  float a3 = v.w * ss[c + 3] + ss[256 + c + 3];
  a0 = a0 / (1.f + __expf(-a0));
  a1 = a1 / (1.f + __expf(-a1));
  a2 = a2 / (1.f + __expf(-a2));
  a3 = a3 / (1.f + __expf(-a3));
  union { uint2 u; __bf16 h[4]; } o;
  o.h[0] = (__bf16)a0; o.h[1] = (__bf16)a1; o.h[2] = (__bf16)a2; o.h[3] = (__bf16)a3;
  *(uint2*)(yb + ((size_t)i << 2)) = o.u;
}

// ---------------- WMMA GEMM, K=256, TDM-staged LDS weight tile ----------------
// Block: 128 M-rows x 64 N-cols (grid.y picks N chunk). B tile (64x256 bf16)
// staged by one 2D TDM descriptor; rows padded 512B->528B (132 dwords,
// 4-bank rotation per row -> conflict-free fragment loads).
enum { EP_SIG = 0, EP_F32 = 1, EP_MUL = 2, EP_OUT = 3 };

template<int EP>
__global__ void __launch_bounds__(256) gemm_k(const __bf16* __restrict__ A,
                                              const __bf16* __restrict__ Bw,
                                              const float*  __restrict__ bias,
                                              const __bf16* __restrict__ gate,
                                              const float*  __restrict__ skip,
                                              void* __restrict__ outv) {
  __shared__ __bf16 sB[64 * 264];                  // 33792 B
  const int tid   = threadIdx.x;
  const int nbase = blockIdx.y * 64;

#if TDM_OK
  if (tid < 32) {
    // group1: data_size=2B, pad_enable, pad_interval=128dw(512B), pad_amount=4dw(16B)
    // tensor_dim0=256, tensor_dim1=64, tile_dim0=256, tile_dim1=64, dim0_stride=256
    i32x8 g1 = {0x07910000, 0x01000000, 0x00400000, 0x01000000, 64, 256, 0, 0};
    i32x4 g2 = {0, 0, 0, 0};
    tdm_load((const char*)Bw + ((size_t)nbase << 9), sB, g1, g2);
    tdm_wait_all();
  }
  __syncthreads();
#else
  {
    const char* gsrc = (const char*)Bw + ((size_t)nbase << 9);  // row = 512B
    for (int q = tid; q < 2048; q += 256) {        // 64 rows x 32 x 16B
      int n = q >> 5, j = q & 31;
      cp16_async((char*)sB + n * 528 + j * 16, gsrc + ((size_t)n << 9) + (j << 4));
    }
  }
  async_wait_all();
  __syncthreads();
#endif

  const int lane  = tid & 31;
  const int wv    = tid >> 5;
  const int mbase = blockIdx.x * 128 + wv * 16;
  const int lm    = lane & 15;
  const int hi    = lane >> 4;
  const int kb    = hi * 8;

  v8f acc[4];
  #pragma unroll
  for (int nf = 0; nf < 4; ++nf) acc[nf] = v8f{0.f,0.f,0.f,0.f,0.f,0.f,0.f,0.f};

  #pragma unroll
  for (int ks = 0; ks < 8; ++ks) {
    v16bf a = ld_frag(A + (((size_t)(mbase + lm)) << 8) + ks * 32 + kb);
    #pragma unroll
    for (int nf = 0; nf < 4; ++nf) {
      v16bf bm = ld_frag((const __bf16*)((const char*)sB +
                  (nf * 16 + lm) * 528 + (ks * 32 + kb) * 2));
      acc[nf] = __builtin_amdgcn_wmma_f32_16x16x32_bf16(false, a, false, bm,
                                                        (short)0, acc[nf], false, false);
    }
  }

  #pragma unroll
  for (int nf = 0; nf < 4; ++nf) {
    const int n = nbase + nf * 16 + lm;
    const float bv = bias[n];
    #pragma unroll
    for (int r = 0; r < 8; ++r) {
      const int m = mbase + r + hi * 8;
      float v = acc[nf][r] + bv;
      if constexpr (EP == EP_SIG) {
        ((__bf16*)outv)[((size_t)m << 8) + n] = (__bf16)(1.f / (1.f + __expf(-v)));
      } else if constexpr (EP == EP_F32) {
        ((float*)outv)[((size_t)m << 8) + n] = v;
      } else if constexpr (EP == EP_MUL) {
        v *= (float)gate[((size_t)m << 8) + n];
        ((__bf16*)outv)[((size_t)m << 8) + n] = (__bf16)v;
      } else { // EP_OUT: 64 out channels, scatter to NCHW with skip, NT store
        const size_t oidx = (((size_t)((m >> 12) * 64 + n)) << 12) + (size_t)(m & 4095);
        __builtin_nontemporal_store(v + skip[oidx], &((float*)outv)[oidx]);
      }
    }
  }
}

// ---------------- chunked EMA scan: s_t = 0.9 s_{t-1} + u_t ----------------
__global__ void scan_chunk_k(const float* __restrict__ u, float* __restrict__ T) {
  int b  = blockIdx.x >> 5;                        // 16 batches
  int ch = blockIdx.x & 31;                        // 32 chunks of 128
  int c  = threadIdx.x;
  const float* p = u + (((size_t)b * LSEQ + ch * 128) << 8) + c;
  float s = 0.f;
  for (int i = 0; i < 128; ++i) {
    __builtin_prefetch(p + ((size_t)(i + 16) << 8), 0, 0);
    s = 0.9f * s + p[(size_t)i << 8];
  }
  T[(((size_t)b * 32 + ch) << 8) + c] = s;
}

__global__ void scan_carry_k(const float* __restrict__ T, float* __restrict__ Carr) {
  int b = blockIdx.x;
  int c = threadIdx.x;
  float p128 = 1.f;
  for (int i = 0; i < 128; ++i) p128 *= 0.9f;      // 0.9^128
  float carry = 0.f;
  for (int ch = 0; ch < 32; ++ch) {
    size_t idx = (((size_t)b * 32 + ch) << 8) + c;
    Carr[idx] = carry;
    carry = T[idx] + p128 * carry;
  }
}

__global__ void scan_final_k(const float* __restrict__ u, const float* __restrict__ Carr,
                             __bf16* __restrict__ st) {
  int b  = blockIdx.x >> 5;
  int ch = blockIdx.x & 31;
  int c  = threadIdx.x;
  const float* p = u + (((size_t)b * LSEQ + ch * 128) << 8) + c;
  __bf16* q = st + (((size_t)b * LSEQ + ch * 128) << 8) + c;
  float s = Carr[(((size_t)b * 32 + ch) << 8) + c];
  for (int i = 0; i < 128; ++i) {
    __builtin_prefetch(p + ((size_t)(i + 16) << 8), 0, 0);
    s = 0.9f * s + p[(size_t)i << 8];
    q[(size_t)i << 8] = (__bf16)s;
  }
}

// ---------------- host orchestration ----------------
extern "C" void kernel_launch(void* const* d_in, const int* in_sizes, int n_in,
                              void* d_out, int out_size, void* d_ws, size_t ws_size,
                              hipStream_t stream) {
  (void)in_sizes; (void)n_in; (void)out_size; (void)ws_size;
  const float* x     = (const float*)d_in[0];
  const float* Wc1   = (const float*)d_in[1];
  const float* bc1   = (const float*)d_in[2];
  const float* gamma = (const float*)d_in[3];
  const float* beta  = (const float*)d_in[4];
  const float* Wg    = (const float*)d_in[5];
  const float* bg    = (const float*)d_in[6];
  const float* Wi    = (const float*)d_in[7];
  const float* bi    = (const float*)d_in[8];
  const float* Ws    = (const float*)d_in[9];
  const float* bs    = (const float*)d_in[10];
  const float* Wo    = (const float*)d_in[11];
  const float* bo    = (const float*)d_in[12];
  float* out = (float*)d_out;

  uint8_t* w = (uint8_t*)d_ws;
  size_t off = 0;
  auto carve = [&](size_t bytes) -> void* {
    off = (off + 255) & ~(size_t)255;
    void* p = w + off;
    off += bytes;
    return p;
  };
  __bf16* xn   = (__bf16*)carve((size_t)B_ * LSEQ * CIN * 2);        // 8.4 MB
  __bf16* wc1p = (__bf16*)carve((size_t)9 * HIDC * CIN * 2);
  __bf16* wgp  = (__bf16*)carve((size_t)HIDC * HIDC * 2);
  __bf16* wip  = (__bf16*)carve((size_t)HIDC * HIDC * 2);
  __bf16* wsp  = (__bf16*)carve((size_t)HIDC * HIDC * 2);
  __bf16* wop  = (__bf16*)carve((size_t)CIN * HIDC * 2);
  float*  ybuf = (float*)carve((size_t)MTOT * HIDC * 4);             // y, later u (67 MB)
  __bf16* yb   = (__bf16*)carve((size_t)MTOT * HIDC * 2);            // act, later states
  __bf16* gate = (__bf16*)carve((size_t)MTOT * HIDC * 2);
  __bf16* xrec = (__bf16*)carve((size_t)MTOT * HIDC * 2);
  float*  Tb   = (float*)carve((size_t)B_ * 32 * HIDC * 4);
  float*  Carr = (float*)carve((size_t)B_ * 32 * HIDC * 4);
  float*  part = (float*)carve((size_t)256 * 2 * HIDC * 4);
  float*  ss   = (float*)carve((size_t)2 * HIDC * 4);

  // prep
  nchw2nhwc_k<<<(B_ * LSEQ * CIN) / 256, 256, 0, stream>>>(x, xn);
  packconv_k<<<(9 * HIDC * CIN) / 256, 256, 0, stream>>>(Wc1, wc1p);
  cast_f32_bf16_k<<<HIDC * HIDC / 256, 256, 0, stream>>>(Wg, wgp, HIDC * HIDC);
  cast_f32_bf16_k<<<HIDC * HIDC / 256, 256, 0, stream>>>(Wi, wip, HIDC * HIDC);
  cast_f32_bf16_k<<<HIDC * HIDC / 256, 256, 0, stream>>>(Ws, wsp, HIDC * HIDC);
  cast_f32_bf16_k<<<CIN * HIDC / 256, 256, 0, stream>>>(Wo, wop, CIN * HIDC);

  // conv + bias -> y (f32, token-major NHWC)
  conv_k<<<dim3(B_ * HH, 4), 256, 0, stream>>>(xn, wc1p, bc1, ybuf);

  // BN + SiLU -> bf16 activations
  bnstats_k<<<256, 256, 0, stream>>>(ybuf, part);
  bnfinal_k<<<1, 256, 0, stream>>>(part, gamma, beta, ss);
  bnact_k<<<(MTOT * HIDC / 4) / 256, 256, 0, stream>>>(ybuf, ss, yb);

  // gate = sigmoid(yb @ Wg^T + bg), u = yb @ Wi^T + bi (u reuses y buffer)
  gemm_k<EP_SIG><<<dim3(MTOT / 128, 4), 256, 0, stream>>>(yb, wgp, bg, nullptr, nullptr, gate);
  gemm_k<EP_F32><<<dim3(MTOT / 128, 4), 256, 0, stream>>>(yb, wip, bi, nullptr, nullptr, ybuf);

  // chunked EMA scan -> states (bf16, reuses yb buffer)
  scan_chunk_k<<<B_ * 32, 256, 0, stream>>>(ybuf, Tb);
  scan_carry_k<<<B_, 256, 0, stream>>>(Tb, Carr);
  scan_final_k<<<B_ * 32, 256, 0, stream>>>(ybuf, Carr, yb);

  // x_rec = gate * (states @ Ws^T + bs)
  gemm_k<EP_MUL><<<dim3(MTOT / 128, 4), 256, 0, stream>>>(yb, wsp, bs, gate, nullptr, xrec);

  // out = skip + NCHW(x_rec @ Wo^T + bo)
  gemm_k<EP_OUT><<<dim3(MTOT / 128, 1), 256, 0, stream>>>(xrec, wop, bo, nullptr, x, out);
}
